// BahdanauAttention_75685913690905
// MI455X (gfx1250) — matrix operand order repaired and checked by reading, not compiled, not run on hardware
//
#include <hip/hip_runtime.h>
#include <stdint.h>

// Problem dims (fixed by the reference)
#define NB    32         // batch
#define NS    4096       // sequence
#define ND    512        // feature dim D
#define NU    128        // hidden dim U
#define SROWS 64         // values rows (s) staged per block
#define LDSK_H 520       // padded bf16 stride per LDS row (512 + 8 pad -> bank spread)

typedef __attribute__((ext_vector_type(16))) __bf16 v16bf;
typedef __attribute__((ext_vector_type(8)))  float  v8f;

union AFrag { v16bf v; uint4 q[2]; };
union BFrag { v16bf v; uint4 q[2]; };
union CFrag { v8f  v; float f[8]; };

__device__ __forceinline__ unsigned pk_bf16(float a, float b) {
#if defined(__AMDGCN__) && __has_builtin(__builtin_amdgcn_cvt_pk_bf16_f32)
  typedef __attribute__((ext_vector_type(2))) __bf16 v2bf;
  v2bf r = __builtin_amdgcn_cvt_pk_bf16_f32(a, b);
  return __builtin_bit_cast(unsigned, r);
#else
  // round-to-nearest-even fp32 -> packed 2x bf16 (a in low half)
  unsigned ua = __float_as_uint(a);
  unsigned ub = __float_as_uint(b);
  ua += 0x7FFFu + ((ua >> 16) & 1u);
  ub += 0x7FFFu + ((ub >> 16) & 1u);
  return (ua >> 16) | (ub & 0xFFFF0000u);
#endif
}

// ---------------------------------------------------------------------------
// Kernel 1: q_proj[b,u] = query[b,:] . W1[:,u] + b1[u]        (32 x 128, tiny)
// ---------------------------------------------------------------------------
__global__ __launch_bounds__(128) void ba_qproj_kernel(
    const float* __restrict__ query, const float* __restrict__ W1,
    const float* __restrict__ b1, float* __restrict__ qproj)
{
  const int b = blockIdx.x, u = threadIdx.x;
  const float* q = query + b * ND;
  float acc = b1[u];
  for (int d = 0; d < ND; ++d) acc += q[d] * W1[d * NU + u];
  qproj[b * NU + u] = acc;
}

// ---------------------------------------------------------------------------
// Kernel 2: repack W2[D,U] fp32 -> w2t[U,D] bf16 (row = u, contiguous K runs)
// ---------------------------------------------------------------------------
__global__ __launch_bounds__(256) void ba_w2t_kernel(
    const float* __restrict__ W2, unsigned short* __restrict__ w2t)
{
  int idx = blockIdx.x * 256 + threadIdx.x;          // 65536 elements
  int u = idx >> 9, d = idx & (ND - 1);
  unsigned ua = __float_as_uint(W2[d * NU + u]);
  ua += 0x7FFFu + ((ua >> 16) & 1u);
  w2t[idx] = (unsigned short)(ua >> 16);
}

// ---------------------------------------------------------------------------
// Kernel 3 (main): per 64-row s-tile, WMMA bf16 GEMM (values @ W2) fused with
// tanh(q_proj + . + b2) . V  ->  unnormalized scores[b,s].
//   block = 256 threads = 8 waves; wave w owns u-tile [16w, 16w+16)
//   values converted fp32->bf16 ONCE at staging; inner loop = ds_load + wmma
// ---------------------------------------------------------------------------
__global__ __launch_bounds__(256) void ba_score_kernel(
    const float* __restrict__ values, const float* __restrict__ b2,
    const float* __restrict__ V, const float* __restrict__ bv,
    const float* __restrict__ qproj, const unsigned short* __restrict__ w2t,
    float* __restrict__ scores)
{
  __shared__ unsigned short lds_vals[SROWS * LDSK_H];   // bf16 tile, padded rows
  __shared__ float lds_score[SROWS];

  const int tile = blockIdx.x;
  const int b  = tile >> 6;                 // NS/SROWS = 64 tiles per batch
  const int s0 = (tile & 63) * SROWS;
  const float* vbase = values + ((size_t)b * NS + s0) * ND;

  const int t    = threadIdx.x;
  const int lane = t & 31;
  const int wave = t >> 5;

  if (t < SROWS) lds_score[t] = 0.0f;

  // ---- stage 64x512 values tile: fp32 global -> bf16 LDS (convert once) ---
#pragma unroll 4
  for (int i = 0; i < 16; ++i) {
    int f8 = i * 256 + t;                   // 8-float chunk index (4096 total)
    int r  = f8 >> 6;                       // 64 chunks per row
    int c8 = f8 & 63;
    const float4* gp = (const float4*)(vbase + (size_t)r * ND + c8 * 8);
    float4 x0 = gp[0], x1 = gp[1];
    uint4 pk;
    pk.x = pk_bf16(x0.x, x0.y); pk.y = pk_bf16(x0.z, x0.w);
    pk.z = pk_bf16(x1.x, x1.y); pk.w = pk_bf16(x1.z, x1.w);
    *(uint4*)&lds_vals[r * LDSK_H + c8 * 8] = pk;
  }
  __syncthreads();

  // ---- preload this wave's 16 B fragments (whole K=512) into registers ----
  // B k order chosen here (contiguous, matches A): lane<16 -> col N=lane,   K 0..15 in V0..7
  //                            lane>=16 -> col N=lane-16, K 16..31 in V0..7
  const int u0    = wave * 16;
  const int ucol  = u0 + (lane & 15);
  const int khalf = (lane < 16) ? 0 : 16;

  BFrag bfr[16];
#pragma unroll
  for (int k = 0; k < 16; ++k) {
    const uint4* p = (const uint4*)(w2t + (size_t)ucol * ND + k * 32 + khalf);
    bfr[k].q[0] = p[0];
    bfr[k].q[1] = p[1];
  }

  const float qb = qproj[b * NU + ucol] + b2[ucol];
  const float Vv = V[ucol];

  // A k order chosen here (contiguous, matches B): lane = row; lane<16 -> K 0..15,
  //                                    lane>=16 -> K 16..31
  const int arow = lane & 15;
  const int kb1  = (lane < 16) ? 0 : 16;
  const int kb2  = kb1 + 8;

#pragma unroll
  for (int mt = 0; mt < 4; ++mt) {
    CFrag c;
#pragma unroll
    for (int j = 0; j < 8; ++j) c.f[j] = 0.0f;

    const unsigned short* rowp = &lds_vals[(mt * 16 + arow) * LDSK_H];
#pragma unroll
    for (int k = 0; k < 16; ++k) {
      AFrag a;
      a.q[0] = *(const uint4*)(rowp + k * 32 + kb1);   // ds_load_b128
      a.q[1] = *(const uint4*)(rowp + k * 32 + kb2);   // ds_load_b128
      c.v = __builtin_amdgcn_wmma_f32_16x16x32_bf16(
          false, a.v, false, bfr[k].v, (short)0, c.v, false, false);
    }

    // epilogue: tanh, multiply by V, reduce the 16 u-columns of this wave
    // C layout: lane<16 -> N=lane, rows j; lane>=16 -> N=lane-16, rows j+8
#pragma unroll
    for (int j = 0; j < 8; ++j) {
      float y = tanhf(c.f[j] + qb) * Vv;
      y += __shfl_xor(y, 1, 32);
      y += __shfl_xor(y, 2, 32);
      y += __shfl_xor(y, 4, 32);
      y += __shfl_xor(y, 8, 32);
      if ((lane & 15) == 0) {
        int r = mt * 16 + j + ((lane < 16) ? 0 : 8);
        atomicAdd(&lds_score[r], y);      // ds_add_f32 across the 8 waves
      }
    }
  }
  __syncthreads();
  if (t < SROWS) scores[(size_t)b * NS + s0 + t] = lds_score[t] + bv[0];
}

// ---------------------------------------------------------------------------
// Kernel 4: softmax over S per batch; writes attention weights to output
// ---------------------------------------------------------------------------
__global__ __launch_bounds__(256) void ba_softmax_kernel(
    const float* __restrict__ scores, float* __restrict__ weights)
{
  __shared__ float red[256];
  const int b = blockIdx.x, t = threadIdx.x;
  const float* s = scores + (size_t)b * NS;

  float m = -3.4e38f;
  for (int i = t; i < NS; i += 256) m = fmaxf(m, s[i]);
  red[t] = m; __syncthreads();
  for (int o = 128; o > 0; o >>= 1) {
    if (t < o) red[t] = fmaxf(red[t], red[t + o]);
    __syncthreads();
  }
  m = red[0]; __syncthreads();

  float sum = 0.0f;
  for (int i = t; i < NS; i += 256) sum += __expf(s[i] - m);
  red[t] = sum; __syncthreads();
  for (int o = 128; o > 0; o >>= 1) {
    if (t < o) red[t] += red[t + o];
    __syncthreads();
  }
  const float inv = 1.0f / red[0];
  for (int i = t; i < NS; i += 256)
    weights[(size_t)b * NS + i] = __expf(s[i] - m) * inv;
}

// ---------------------------------------------------------------------------
// Kernel 5: partial context per (b, s-chunk): 8 chunks of 512 rows
//   partial[ch][b][d] = sum_{s in chunk} w[b,s] * values[b,s,d]
//   weight chunk async-staged into LDS (CDNA5 global->LDS async path)
// ---------------------------------------------------------------------------
__global__ __launch_bounds__(128) void ba_ctx_partial_kernel(
    const float* __restrict__ values, const float* __restrict__ weights,
    float* __restrict__ partial)
{
  __shared__ float wsm[512];
  const int b = blockIdx.x, ch = blockIdx.y, t = threadIdx.x;  // t: float4 col
  const int srow = ch * 512;
  const float* w = weights + (size_t)b * NS + srow;

  // async-stage the 512 weights for this chunk into LDS (1 b128 per lane)
  {
    unsigned ldsa = (unsigned)(uintptr_t)(&wsm[t * 4]);
    uint64_t ga   = (uint64_t)(uintptr_t)(w + t * 4);
    asm volatile("global_load_async_to_lds_b128 %0, %1, off"
                 :: "v"(ldsa), "v"(ga) : "memory");
    asm volatile("s_wait_asynccnt 0" ::: "memory");
  }
  __syncthreads();

  const float4* vp = (const float4*)(values + ((size_t)b * NS + srow) * ND) + t;
  float4 acc = make_float4(0.f, 0.f, 0.f, 0.f);
#pragma unroll 4
  for (int s = 0; s < 512; ++s) {
    float  wv = wsm[s];
    float4 v  = vp[(size_t)s * (ND / 4)];
    acc.x += wv * v.x; acc.y += wv * v.y; acc.z += wv * v.z; acc.w += wv * v.w;
  }
  ((float4*)(partial + ((size_t)(ch * NB + b)) * ND))[t] = acc;
}

// ---------------------------------------------------------------------------
// Kernel 6: deterministic reduction of the 8 partial contexts
// ---------------------------------------------------------------------------
__global__ __launch_bounds__(256) void ba_ctx_reduce_kernel(
    const float* __restrict__ partial, float* __restrict__ out_ctx)
{
  int i = blockIdx.x * 256 + threadIdx.x;   // 16384 = NB*ND
  float s = 0.0f;
#pragma unroll
  for (int ch = 0; ch < 8; ++ch) s += partial[(size_t)ch * (NB * ND) + i];
  out_ctx[i] = s;
}

// ---------------------------------------------------------------------------
extern "C" void kernel_launch(void* const* d_in, const int* in_sizes, int n_in,
                              void* d_out, int out_size, void* d_ws, size_t ws_size,
                              hipStream_t stream) {
  const float* query  = (const float*)d_in[0];   // [32,512]
  const float* values = (const float*)d_in[1];   // [32,4096,512]
  const float* W1     = (const float*)d_in[2];   // [512,128]
  const float* b1     = (const float*)d_in[3];   // [128]
  const float* W2     = (const float*)d_in[4];   // [512,128]
  const float* b2     = (const float*)d_in[5];   // [128]
  const float* V      = (const float*)d_in[6];   // [128,1]
  const float* bv     = (const float*)d_in[7];   // [1]

  float* out_ctx = (float*)d_out;                // [32,512]
  float* out_w   = out_ctx + NB * ND;            // [32,4096]

  // workspace layout (~1.15 MB total)
  float*          qproj   = (float*)d_ws;                    // 4096 f32
  float*          scores  = qproj + NB * NU;                 // 131072 f32
  float*          partial = scores + NB * NS;                // 131072 f32
  unsigned short* w2t     = (unsigned short*)(partial + NB * ND * 8); // 65536 bf16

  ba_qproj_kernel<<<NB, NU, 0, stream>>>(query, W1, b1, qproj);
  ba_w2t_kernel<<<(NU * ND) / 256, 256, 0, stream>>>(W2, w2t);
  ba_score_kernel<<<(NB * NS) / SROWS, 256, 0, stream>>>(
      values, b2, V, bv, qproj, w2t, scores);
  ba_softmax_kernel<<<NB, 256, 0, stream>>>(scores, out_w);
  ba_ctx_partial_kernel<<<dim3(NB, 8), 128, 0, stream>>>(values, out_w, partial);
  ba_ctx_reduce_kernel<<<(NB * ND) / 256, 256, 0, stream>>>(partial, out_ctx);
}
